// MLSTMFixD_62354335203968
// MI455X (gfx1250) — compile-verified, matrix operand order checked by reading, not run on hardware
//
#include <hip/hip_runtime.h>
#include <hip/hip_bf16.h>
#include <math.h>

// Problem dims (fixed by reference setup_inputs)
#define Tn 128
#define Bn 256
#define In 512
#define Hn 512
#define KH 32
#define On 5

typedef __attribute__((ext_vector_type(16))) __bf16 v16bf;
typedef __attribute__((ext_vector_type(8)))  float  v8f;
typedef __attribute__((ext_vector_type(4)))  int    v4i_;

struct __attribute__((aligned(16))) U128 { unsigned int x, y, z, w; };
struct __attribute__((aligned(32))) U256 { U128 lo, hi; };

#if defined(__gfx1250__) && __has_builtin(__builtin_amdgcn_global_load_async_to_lds_b128)
#define HAVE_ASYNC_LDS 1
typedef __attribute__((address_space(1))) v4i_* g_v4i_p;
typedef __attribute__((address_space(3))) v4i_* l_v4i_p;
__device__ __forceinline__ void async_ld16(const unsigned short* gp, unsigned short* lp) {
  // DMA 16B/lane global -> LDS, tracked by ASYNCcnt (no VGPR data movement)
  __builtin_amdgcn_global_load_async_to_lds_b128((g_v4i_p)gp, (l_v4i_p)lp, 0, 0);
}
__device__ __forceinline__ void wait_async0() {
#if __has_builtin(__builtin_amdgcn_s_wait_asynccnt)
  __builtin_amdgcn_s_wait_asynccnt(0);
#else
  asm volatile("s_wait_asynccnt 0x0" ::: "memory");
#endif
}
#else
#define HAVE_ASYNC_LDS 0
#endif

__device__ __forceinline__ unsigned short f2bf(float f) {
  unsigned int u = __float_as_uint(f);
  u += 0x7FFFu + ((u >> 16) & 1u);   // round-to-nearest-even
  return (unsigned short)(u >> 16);
}
__device__ __forceinline__ float sigmoidf_(float x) { return 1.0f / (1.0f + expf(-x)); }

// ---------------- setup kernels ----------------

__global__ void cvt_f32_bf16(const float* __restrict__ src,
                             unsigned short* __restrict__ dst, int n) {
  int i = (blockIdx.x * blockDim.x + threadIdx.x) * 4;
  if (i + 3 < n) {
    float4 v = *(const float4*)(src + i);
    dst[i + 0] = f2bf(v.x);
    dst[i + 1] = f2bf(v.y);
    dst[i + 2] = f2bf(v.z);
    dst[i + 3] = f2bf(v.w);
  } else {
    for (; i < n; ++i) dst[i] = f2bf(src[i]);
  }
}

// Pack [Wa | Wb] (each rows x 512, row-major f32) -> rows x 1024 bf16 row-major
__global__ void pack_w(const float* __restrict__ Wa, const float* __restrict__ Wb,
                       unsigned short* __restrict__ dst, int rows) {
  int i = blockIdx.x * blockDim.x + threadIdx.x;
  if (i >= rows * 1024) return;
  int r = i >> 10, c = i & 1023;
  float v = (c < 512) ? Wa[r * 512 + c] : Wb[r * 512 + (c - 512)];
  dst[i] = f2bf(v);
}

// Folded biases + fractional-differencing coefficients (oldest-lag-first).
__global__ void setup_small(const float* __restrict__ b_d,
                            const float* __restrict__ b_mih, const float* __restrict__ b_mhh,
                            const float* __restrict__ b_ih,  const float* __restrict__ b_hh,
                            float* __restrict__ wd, float* __restrict__ bias_m,
                            float* __restrict__ bias_l) {
  int i = blockIdx.x * blockDim.x + threadIdx.x;
  if (i < 3 * Hn) bias_m[i] = b_mih[i] + b_mhh[i];
  if (i < 4 * Hn) bias_l[i] = b_ih[i] + b_hh[i];
  if (i < Hn) {
    float d = 0.5f * sigmoidf_(b_d[i]);
    float c = 1.0f;
    for (int j = 0; j < KH; ++j) {
      c *= (((float)j) - d) / ((float)j + 1.0f);  // cumprod -> c_{j+1}
      wd[(KH - 1 - j) * Hn + i] = c;              // reversed: wd[s] = coeff[K-1-s]
    }
  }
}

// ---------------- WMMA GEMM: C(MxN) = [A0|A1](Mx1024 bf16) * W^T (Nx1024 bf16) ----------------
// Block tile 64(M) x 128(N), 8 waves, each wave 16x64 via 4 accumulators, K-chunk 32.
// Async-DMA double-buffered LDS staging when the toolchain exposes the gfx1250 builtin.
__global__ __launch_bounds__(256) void gemm_rec(
    const unsigned short* __restrict__ A0,  // M x 512 bf16 row-major (K = 0..511)
    const unsigned short* __restrict__ A1,  // M x 512 bf16 row-major (K = 512..1023)
    const unsigned short* __restrict__ W,   // N x 1024 bf16 row-major
    float* __restrict__ C,                  // M x N f32
    int N) {
#if HAVE_ASYNC_LDS
  __shared__ unsigned short As[2][64 * 48];   // +16-half pad keeps 16B alignment
  __shared__ unsigned short Ws[2][128 * 48];
#else
  __shared__ unsigned short As[1][64 * 48];
  __shared__ unsigned short Ws[1][128 * 48];
#endif

  const int tid   = threadIdx.x;
  const int lane  = tid & 31;
  const int wv    = tid >> 5;
  const int mBase = blockIdx.y * 64;
  const int nBase = blockIdx.x * 128;
  const int m0 = (wv & 3) * 16;
  const int n0 = (wv >> 2) * 64;
  const int fr   = lane & 15;              // fragment row within 16
  const int bofs = (lane < 16) ? 0 : 8;    // ISA 16-bit A/B layout K-group base

  v8f zero = {0.f, 0.f, 0.f, 0.f, 0.f, 0.f, 0.f, 0.f};
  v8f acc[4] = {zero, zero, zero, zero};

  const int arow = tid >> 2;               // 0..63
  const int acol = (tid & 3) << 3;         // 0,8,16,24 halves
  const int wrow = tid >> 1;               // 0..127
  const int wcol = (tid & 1) << 4;         // 0,16 halves

  auto compute_chunk = [&](const unsigned short* Asb, const unsigned short* Wsb) {
    U256 ta;
    ta.lo = *(const U128*)&Asb[(m0 + fr) * 48 + bofs];
    ta.hi = *(const U128*)&Asb[(m0 + fr) * 48 + bofs + 16];
    v16bf afrag = __builtin_bit_cast(v16bf, ta);
#pragma unroll
    for (int j = 0; j < 4; ++j) {
      U256 tw;
      tw.lo = *(const U128*)&Wsb[(n0 + j * 16 + fr) * 48 + bofs];
      tw.hi = *(const U128*)&Wsb[(n0 + j * 16 + fr) * 48 + bofs + 16];
      v16bf wfrag = __builtin_bit_cast(v16bf, tw);
      acc[j] = __builtin_amdgcn_wmma_f32_16x16x32_bf16(
          false, afrag, false, wfrag, (short)0, acc[j], false, false);
    }
  };

#if HAVE_ASYNC_LDS
  auto issue_chunk = [&](int buf, int kc) {
    const unsigned short* Aseg = (kc < 16) ? A0 : A1;
    const int kseg = (kc & 15) << 5;
    async_ld16(&Aseg[(mBase + arow) * 512 + kseg + acol], &As[buf][arow * 48 + acol]);
    const unsigned short* wp = &W[(nBase + wrow) * 1024 + (kc << 5) + wcol];
    async_ld16(wp,     &Ws[buf][wrow * 48 + wcol]);
    async_ld16(wp + 8, &Ws[buf][wrow * 48 + wcol + 8]);
  };

  issue_chunk(0, 0);
  for (int kc = 0; kc < 32; ++kc) {
    const int cur = kc & 1;
    wait_async0();        // current buffer's DMA has landed in LDS
    __syncthreads();      // publish across waves; prior reads of other buffer retired
    if (kc + 1 < 32) issue_chunk(cur ^ 1, kc + 1);  // overlap DMA with WMMA below
    compute_chunk(As[cur], Ws[cur]);
  }
#else
  for (int kc = 0; kc < 32; ++kc) {
    const unsigned short* Aseg = (kc < 16) ? A0 : A1;
    const int kseg = (kc & 15) << 5;

    __syncthreads();
    *(U128*)&As[0][arow * 48 + acol] =
        *(const U128*)&Aseg[(mBase + arow) * 512 + kseg + acol];
    const unsigned short* wp = &W[(nBase + wrow) * 1024 + (kc << 5) + wcol];
    *(U128*)&Ws[0][wrow * 48 + wcol]     = *(const U128*)(wp);
    *(U128*)&Ws[0][wrow * 48 + wcol + 8] = *(const U128*)(wp + 8);
    if (kc + 1 < 32) {
      __builtin_prefetch(&W[(nBase + wrow) * 1024 + ((kc + 1) << 5) + wcol], 0, 0);
    }
    __syncthreads();
    compute_chunk(As[0], Ws[0]);
  }
#endif

  // D layout: lane holds N = lane&15, M = r + 8*(lane>>4)
  const int mrow = mBase + m0 + ((lane >> 4) << 3);
#pragma unroll
  for (int j = 0; j < 4; ++j) {
    const int col = nBase + n0 + j * 16 + fr;
#pragma unroll
    for (int r = 0; r < 8; ++r) {
      C[(size_t)(mrow + r) * N + col] = acc[j][r];
    }
  }
}

// ---------------- elementwise cell kernels ----------------

// mLSTM gates + fractional-d long-memory filter over K=32 cell-state ring.
// Ring slot (t+s)%K holds hist[s] (s=0 oldest); new c1 overwrites slot t%K.
__global__ void cell_mlstm(const float* __restrict__ g, const float* __restrict__ bias_m,
                           const float* __restrict__ wd, float* __restrict__ R,
                           unsigned short* __restrict__ h1bf, int t) {
  int idx = blockIdx.x * blockDim.x + threadIdx.x;   // over B*H
  int b = idx >> 9, h = idx & 511;
  float ig = g[b * 1536 + h]        + bias_m[h];
  float og = g[b * 1536 + 512 + h]  + bias_m[512 + h];
  float gg = g[b * 1536 + 1024 + h] + bias_m[1024 + h];
  float i_g = sigmoidf_(ig), o_g = sigmoidf_(og), cand = tanhf(gg);
  float acc = 0.f;
#pragma unroll
  for (int s = 0; s < KH; ++s)
    acc += wd[s * Hn + h] * R[(size_t)((t + s) & (KH - 1)) * (Bn * Hn) + idx];
  float c1 = i_g * cand - acc;
  R[(size_t)(t & (KH - 1)) * (Bn * Hn) + idx] = c1;
  h1bf[idx] = f2bf(o_g * tanhf(c1));
}

__global__ void cell_lstm(const float* __restrict__ g2, const float* __restrict__ bias_l,
                          float* __restrict__ c2, unsigned short* __restrict__ h2bf,
                          float* __restrict__ sel, const int* __restrict__ length, int t) {
  int idx = blockIdx.x * blockDim.x + threadIdx.x;   // over B*H
  int b = idx >> 9, h = idx & 511;
  float i2 = g2[b * 2048 + h]        + bias_l[h];
  float f2 = g2[b * 2048 + 512 + h]  + bias_l[512 + h];
  float gc = g2[b * 2048 + 1024 + h] + bias_l[1024 + h];
  float o2 = g2[b * 2048 + 1536 + h] + bias_l[1536 + h];
  float c2n = sigmoidf_(f2) * c2[idx] + sigmoidf_(i2) * tanhf(gc);
  c2[idx] = c2n;
  float h2 = tanhf(sigmoidf_(o2) * tanhf(c2n));  // extra tanh as in source
  h2bf[idx] = f2bf(h2);
  if (length[b] == t) sel[idx] = h2;             // outs[length[b], b]
}

// ---------------- head: logits + log_softmax ----------------
__global__ void head_logsoftmax(const float* __restrict__ sel, const float* __restrict__ Wout,
                                const float* __restrict__ bout, float* __restrict__ out) {
  __shared__ float part[On][64];
  int b = blockIdx.x, tid = threadIdx.x;
#pragma unroll
  for (int o = 0; o < On; ++o) {
    float s = 0.f;
    for (int h = tid; h < Hn; h += 64) s += sel[b * Hn + h] * Wout[o * Hn + h];
    part[o][tid] = s;
  }
  __syncthreads();
  if (tid == 0) {
    float l[On];
    float m = -1e30f;
    for (int o = 0; o < On; ++o) {
      float s = bout[o];
      for (int i = 0; i < 64; ++i) s += part[o][i];
      l[o] = s;
      m = fmaxf(m, s);
    }
    float lse = 0.f;
    for (int o = 0; o < On; ++o) lse += expf(l[o] - m);
    lse = logf(lse);
    for (int o = 0; o < On; ++o) out[b * On + o] = l[o] - m - lse;
  }
}

// ---------------- host launcher ----------------

extern "C" void kernel_launch(void* const* d_in, const int* in_sizes, int n_in,
                              void* d_out, int out_size, void* d_ws, size_t ws_size,
                              hipStream_t stream) {
  (void)in_sizes; (void)n_in; (void)out_size; (void)ws_size;
  const float* x      = (const float*)d_in[0];
  const int*   length = (const int*)d_in[1];
  const float* b_d    = (const float*)d_in[2];
  const float* W_mih  = (const float*)d_in[3];
  const float* W_mhh  = (const float*)d_in[4];
  const float* b_mih  = (const float*)d_in[5];
  const float* b_mhh  = (const float*)d_in[6];
  const float* W_ih   = (const float*)d_in[7];
  const float* W_hh   = (const float*)d_in[8];
  const float* b_ih   = (const float*)d_in[9];
  const float* b_hh   = (const float*)d_in[10];
  const float* W_out  = (const float*)d_in[11];
  const float* b_out  = (const float*)d_in[12];
  float* out = (float*)d_out;

  // Workspace carve-out (~63 MB), 256B-aligned regions.
  char* ws = (char*)d_ws;
  size_t off = 0;
  auto alloc = [&](size_t bytes) {
    char* p = ws + off;
    off = (off + bytes + 255) & ~(size_t)255;
    return p;
  };
  unsigned short* Xbf  = (unsigned short*)alloc((size_t)Tn * Bn * In * 2);
  unsigned short* Wm   = (unsigned short*)alloc((size_t)3 * Hn * 1024 * 2);
  unsigned short* Wl   = (unsigned short*)alloc((size_t)4 * Hn * 1024 * 2);
  unsigned short* h1bf = (unsigned short*)alloc((size_t)Bn * Hn * 2);
  unsigned short* h2bf = (unsigned short*)alloc((size_t)Bn * Hn * 2);
  float* g      = (float*)alloc((size_t)Bn * 3 * Hn * 4);
  float* g2     = (float*)alloc((size_t)Bn * 4 * Hn * 4);
  float* R      = (float*)alloc((size_t)KH * Bn * Hn * 4);
  float* c2     = (float*)alloc((size_t)Bn * Hn * 4);
  float* sel    = (float*)alloc((size_t)Bn * Hn * 4);
  float* wd     = (float*)alloc((size_t)KH * Hn * 4);
  float* bias_m = (float*)alloc((size_t)3 * Hn * 4);
  float* bias_l = (float*)alloc((size_t)4 * Hn * 4);

  (void)hipMemsetAsync(R,    0, (size_t)KH * Bn * Hn * 4, stream);
  (void)hipMemsetAsync(h1bf, 0, (size_t)Bn * Hn * 2, stream);
  (void)hipMemsetAsync(h2bf, 0, (size_t)Bn * Hn * 2, stream);
  (void)hipMemsetAsync(c2,   0, (size_t)Bn * Hn * 4, stream);

  const int nx = Tn * Bn * In;
  cvt_f32_bf16<<<(nx / 4 + 255) / 256, 256, 0, stream>>>(x, Xbf, nx);
  pack_w<<<(3 * Hn * 1024 + 255) / 256, 256, 0, stream>>>(W_mih, W_mhh, Wm, 3 * Hn);
  pack_w<<<(4 * Hn * 1024 + 255) / 256, 256, 0, stream>>>(W_ih, W_hh, Wl, 4 * Hn);
  setup_small<<<8, 256, 0, stream>>>(b_d, b_mih, b_mhh, b_ih, b_hh, wd, bias_m, bias_l);

  for (int t = 0; t < Tn; ++t) {
    // g = [x_t | h1] @ [W_mih | W_mhh]^T   (B x 3H)
    gemm_rec<<<dim3((3 * Hn) / 128, Bn / 64), 256, 0, stream>>>(
        Xbf + (size_t)t * Bn * In, h1bf, Wm, g, 3 * Hn);
    cell_mlstm<<<(Bn * Hn) / 256, 256, 0, stream>>>(g, bias_m, wd, R, h1bf, t);
    // g2 = [h1n | h2] @ [W_ih | W_hh]^T    (B x 4H)
    gemm_rec<<<dim3((4 * Hn) / 128, Bn / 64), 256, 0, stream>>>(
        h1bf, h2bf, Wl, g2, 4 * Hn);
    cell_lstm<<<(Bn * Hn) / 256, 256, 0, stream>>>(g2, bias_l, c2, h2bf, sel, length, t);
  }
  head_logsoftmax<<<Bn, 64, 0, stream>>>(sel, W_out, b_out, out);
}
